// PointNetPlusPlusAttentionSeg_86165633892453
// MI455X (gfx1250) — compile-verified, matrix operand order checked by reading, not run on hardware
//
#include <hip/hip_runtime.h>
#include <math.h>

// ---------------------------------------------------------------------------
// PointNet++ attention segmentation for MI455X (gfx1250, wave32, WMMA).
// - All conv layers  -> k_wmma_gemm  (V_WMMA_F32_16X16X32_F16, f32 accum),
//   branch-free hot loop, float4 A loads, coalesced B loads, prefetch.
// - position_attention -> k_flash_pos_attn (fused QK^T + pos + online softmax
//   + P*V). Never materializes the 4x4096x4096 attention matrix (saves ~1GB
//   of HBM traffic at 23.3 TB/s). V tile staged through LDS (f16, coalesced).
// - fuse layer's broadcast channels folded into a per-batch column bias
//   (avoids a 92MB concat buffer).
// ---------------------------------------------------------------------------

#define BN_INV_C 0.9999950000374997f  // 1/sqrt(1+1e-5)

typedef __attribute__((ext_vector_type(16))) _Float16 v16h;
typedef __attribute__((ext_vector_type(8)))  float    v8f;

__device__ __forceinline__ int kmap(int j, int hf) {
  // 16-bit A/B fragment k-index for element j of a lane (CDNA5 ISA 7.12.2):
  // j=0..7 -> k = hf*8 + j ; j=8..15 -> k = 16 + hf*8 + (j-8)
  return ((j >> 3) << 4) + (hf << 3) + (j & 7);
}

// ---------------------------------------------------------------------------
// Generic WMMA GEMM:  C[M,N] = epilogue( A[M,K] * B[K,N] )
// A row-major, leading dim lda; B row-major (K x N); C row-major (M x N).
// flags: 1=+bias[m], 2=batchnorm(g,be), 4=relu.
// cbias: per-batch column bias cbias[(n/cb_div)*M + m].
// One wave per 16x16 tile. Hot loop is branch-free: addresses are clamped
// in-bounds, out-of-range lanes only masked at the epilogue store; the K tail
// uses value selects (cndmask), not exec-mask branches.
// ---------------------------------------------------------------------------
__global__ void k_wmma_gemm(const float* __restrict__ A, const float* __restrict__ Bm,
                            const float* __restrict__ bias, const float* __restrict__ gg,
                            const float* __restrict__ bbe, const float* __restrict__ cbias,
                            float* __restrict__ C,
                            int M, int N, int K, int lda, int cb_div, int flags)
{
  const int lane = threadIdx.x & 31;
  const int hf   = lane >> 4;
  const int lm   = lane & 15;
  const int n0   = blockIdx.x * 16;
  const int m0   = blockIdx.y * 16;

  const int am   = min(m0 + lm, M - 1);   // clamped: loads always in-bounds
  const int bn_r = n0 + lm;
  const int bn   = min(bn_r, N - 1);
  const float* __restrict__ Arow = A + (long long)am * lda;
  const float* __restrict__ Bcol = Bm + bn;

  v8f acc = {};
  const int Kfull = K & ~31;

  if ((lda & 3) == 0) {
    // fast path: 16B-aligned A rows -> four global_load_b128 per chunk
    for (int k0 = 0; k0 < Kfull; k0 += 32) {
      __builtin_prefetch(Bcol + (long long)min(k0 + 32, K - 1) * N, 0, 1);
      const float4 a0 = *(const float4*)(Arow + k0 + hf * 8);
      const float4 a1 = *(const float4*)(Arow + k0 + hf * 8 + 4);
      const float4 a2 = *(const float4*)(Arow + k0 + 16 + hf * 8);
      const float4 a3 = *(const float4*)(Arow + k0 + 16 + hf * 8 + 4);
      v16h af, bf;
      af[0] = (_Float16)a0.x; af[1] = (_Float16)a0.y; af[2]  = (_Float16)a0.z; af[3]  = (_Float16)a0.w;
      af[4] = (_Float16)a1.x; af[5] = (_Float16)a1.y; af[6]  = (_Float16)a1.z; af[7]  = (_Float16)a1.w;
      af[8] = (_Float16)a2.x; af[9] = (_Float16)a2.y; af[10] = (_Float16)a2.z; af[11] = (_Float16)a2.w;
      af[12] = (_Float16)a3.x; af[13] = (_Float16)a3.y; af[14] = (_Float16)a3.z; af[15] = (_Float16)a3.w;
#pragma unroll
      for (int j = 0; j < 16; ++j)
        bf[j] = (_Float16)Bcol[(long long)(k0 + kmap(j, hf)) * N];
      acc = __builtin_amdgcn_wmma_f32_16x16x32_f16(false, af, false, bf,
                                                   (short)0, acc, false, false);
    }
  } else {
    for (int k0 = 0; k0 < Kfull; k0 += 32) {
      v16h af, bf;
#pragma unroll
      for (int j = 0; j < 16; ++j) {
        int k = k0 + kmap(j, hf);
        af[j] = (_Float16)Arow[k];
        bf[j] = (_Float16)Bcol[(long long)k * N];
      }
      acc = __builtin_amdgcn_wmma_f32_16x16x32_f16(false, af, false, bf,
                                                   (short)0, acc, false, false);
    }
  }

  if (Kfull < K) {  // tail chunk: value-select zeros, no exec branches
    v16h af, bf;
#pragma unroll
    for (int j = 0; j < 16; ++j) {
      int k  = Kfull + kmap(j, hf);
      int kc = min(k, K - 1);
      float av = Arow[kc];
      float bv = Bcol[(long long)kc * N];
      af[j] = (_Float16)((k < K) ? av : 0.f);
      bf[j] = (_Float16)((k < K) ? bv : 0.f);
    }
    acc = __builtin_amdgcn_wmma_f32_16x16x32_f16(false, af, false, bf,
                                                 (short)0, acc, false, false);
  }

  if (bn_r >= N) return;
#pragma unroll
  for (int v = 0; v < 8; ++v) {
    int m = m0 + v + 8 * hf;
    if (m >= M) continue;
    float y = acc[v];
    if (flags & 1) y += bias[m];
    if (cbias)     y += cbias[(long long)(bn_r / cb_div) * M + m];
    if (flags & 2) y = gg[m] * (y * BN_INV_C) + bbe[m];
    if (flags & 4) y = fmaxf(y, 0.f);
    C[(long long)m * N + bn_r] = y;
  }
}

// ---------------------------------------------------------------------------
// Fused position attention (flash style). One wave handles 16 query rows.
// Q,K: (32 x ldN), V,xin,out: (128 x ldN), column = b*N + n.
// out = gamma * softmax(Q^T K + 0.1*pos) V^T + xin
// V tile (128ch x 32m) staged through LDS as f16 with coalesced 128B rows.
// ---------------------------------------------------------------------------
__global__ void k_flash_pos_attn(const float* __restrict__ Q, const float* __restrict__ Kt,
                                 const float* __restrict__ V, const float* __restrict__ xyzn,
                                 const float* __restrict__ xin, const float* __restrict__ gamma,
                                 float* __restrict__ out, int N, int ldN)
{
  const int lane = threadIdx.x & 31;
  const int hf   = lane >> 4;
  const int lm   = lane & 15;
  const int b    = blockIdx.y;
  const int n0   = blockIdx.x * 16;
  const long long base = (long long)b * N;

  __shared__ float    sS[16 * 32];
  __shared__ float    sMax[16], sSum[16], sAlpha[16];
  __shared__ float    sXn[16][3];
  __shared__ _Float16 sV[128][32];

  if (lane < 16) {
    sMax[lane] = -1e30f;
    sSum[lane] = 0.f;
    long long p = (base + n0 + lane) * 3;
    sXn[lane][0] = xyzn[p]; sXn[lane][1] = xyzn[p + 1]; sXn[lane][2] = xyzn[p + 2];
  }

  // Q fragment (A operand, 16 rows x 32 channels), loaded once
  v16h aq;
#pragma unroll
  for (int j = 0; j < 16; ++j) {
    int cch = kmap(j, hf);
    aq[j] = (_Float16)Q[(long long)cch * ldN + base + n0 + lm];
  }

  v8f o[8];
#pragma unroll
  for (int cg = 0; cg < 8; ++cg) o[cg] = (v8f){};

  __syncthreads();

  for (int m0 = 0; m0 < N; m0 += 32) {
    // ---- stage V tile: 128 rows x 32 cols, coalesced 128B per row ----
    {
      const float* vp = V + base + m0 + lane;
#pragma unroll 8
      for (int r = 0; r < 128; ++r)
        sV[r][lane] = (_Float16)vp[(long long)r * ldN];
    }

    // ---- S tile (16 x 32) = Q^T K via two WMMAs ----
#pragma unroll
    for (int t = 0; t < 2; ++t) {
      v16h bk;
#pragma unroll
      for (int j = 0; j < 16; ++j) {
        int cch = kmap(j, hf);
        bk[j] = (_Float16)Kt[(long long)cch * ldN + base + m0 + t * 16 + lm];
      }
      v8f zz = {};
      v8f s = __builtin_amdgcn_wmma_f32_16x16x32_f16(false, aq, false, bk,
                                                     (short)0, zz, false, false);
#pragma unroll
      for (int v = 0; v < 8; ++v)
        sS[(v + 8 * hf) * 32 + t * 16 + lm] = s[v];
    }
    __syncthreads();

    // ---- + 0.1 * pos (dot of normalized xyz) ----
    {
      long long mp = (base + m0 + lane) * 3;
      float xm0 = xyzn[mp], xm1 = xyzn[mp + 1], xm2 = xyzn[mp + 2];
      for (int r = 0; r < 16; ++r) {
        float d = sXn[r][0] * xm0 + sXn[r][1] * xm1 + sXn[r][2] * xm2;
        sS[r * 32 + lane] += 0.1f * d;
      }
    }
    __syncthreads();

    // ---- online softmax bookkeeping (lanes 0..15, one row each) ----
    if (lane < 16) {
      float mOld = sMax[lane];
      float mNew = mOld;
      for (int c = 0; c < 32; ++c) mNew = fmaxf(mNew, sS[lane * 32 + c]);
      float alpha = expf(mOld - mNew);
      float ssum = 0.f;
      for (int c = 0; c < 32; ++c) {
        float p = expf(sS[lane * 32 + c] - mNew);
        sS[lane * 32 + c] = p;
        ssum += p;
      }
      sMax[lane] = mNew;
      sSum[lane] = sSum[lane] * alpha + ssum;
      sAlpha[lane] = alpha;
    }
    __syncthreads();

    // ---- rescale accumulators by per-row alpha ----
#pragma unroll
    for (int cg = 0; cg < 8; ++cg)
#pragma unroll
      for (int v = 0; v < 8; ++v)
        o[cg][v] *= sAlpha[v + 8 * hf];

    // ---- O += P * V^T  (8 channel groups of 16), V from LDS ----
    v16h ap;
#pragma unroll
    for (int j = 0; j < 16; ++j)
      ap[j] = (_Float16)sS[lm * 32 + kmap(j, hf)];
#pragma unroll
    for (int cg = 0; cg < 8; ++cg) {
      v16h bv;
#pragma unroll
      for (int j = 0; j < 16; ++j)
        bv[j] = sV[cg * 16 + lm][kmap(j, hf)];
      o[cg] = __builtin_amdgcn_wmma_f32_16x16x32_f16(false, ap, false, bv,
                                                     (short)0, o[cg], false, false);
    }
    __syncthreads();
  }

  const float gm = gamma[0];
#pragma unroll
  for (int cg = 0; cg < 8; ++cg) {
    int cch = cg * 16 + lm;
#pragma unroll
    for (int v = 0; v < 8; ++v) {
      int nr = v + 8 * hf;
      float y = o[cg][v] / sSum[nr];
      long long idx = (long long)cch * ldN + base + n0 + nr;
      out[idx] = gm * y + xin[idx];
    }
  }
}

// ------------------------- scalar helper kernels ---------------------------

__global__ void k_prep(const float* __restrict__ x, float* feat0, float* xyz,
                       float* xyzn, int BN)
{
  int i = blockIdx.x * blockDim.x + threadIdx.x;
  if (i >= BN) return;
  float v[6];
  for (int c = 0; c < 6; ++c) { v[c] = x[(long long)i * 6 + c]; feat0[(long long)c * BN + i] = v[c]; }
  xyz[(long long)i * 3] = v[0]; xyz[(long long)i * 3 + 1] = v[1]; xyz[(long long)i * 3 + 2] = v[2];
  float nr = sqrtf(v[0] * v[0] + v[1] * v[1] + v[2] * v[2]) + 1e-8f;
  xyzn[(long long)i * 3] = v[0] / nr; xyzn[(long long)i * 3 + 1] = v[1] / nr; xyzn[(long long)i * 3 + 2] = v[2] / nr;
}

__global__ void k_fps(const float* __restrict__ xyz, int Nl, int npoint, int* __restrict__ out)
{
  int b = blockIdx.x;
  int tid = threadIdx.x;
  __shared__ float sd[4096];
  __shared__ float rv[256];
  __shared__ int   ri[256];
  __shared__ int   sFar;
  for (int i = tid; i < Nl; i += 256) sd[i] = 1e30f;
  if (tid == 0) sFar = Nl / 2;
  __syncthreads();
  for (int it = 0; it < npoint; ++it) {
    int far = sFar;
    if (tid == 0) out[b * npoint + it] = far;
    long long cb = ((long long)b * Nl + far) * 3;
    float cx = xyz[cb], cy = xyz[cb + 1], cz = xyz[cb + 2];
    float bestv = -1.f; int besti = 0;
    for (int j = tid; j < Nl; j += 256) {
      long long q = ((long long)b * Nl + j) * 3;
      float dx = xyz[q] - cx, dy = xyz[q + 1] - cy, dz = xyz[q + 2] - cz;
      float nd = fminf(sd[j], dx * dx + dy * dy + dz * dz);
      sd[j] = nd;
      if (nd > bestv) { bestv = nd; besti = j; }
    }
    rv[tid] = bestv; ri[tid] = besti;
    __syncthreads();
    for (int s = 128; s > 0; s >>= 1) {
      if (tid < s) {
        if (rv[tid + s] > rv[tid] || (rv[tid + s] == rv[tid] && ri[tid + s] < ri[tid])) {
          rv[tid] = rv[tid + s]; ri[tid] = ri[tid + s];
        }
      }
      __syncthreads();
    }
    if (tid == 0) sFar = ri[0];
    __syncthreads();
  }
}

__global__ void k_gather_xyz(const float* __restrict__ xyz, int Nl,
                             const int* __restrict__ idx, int np, float* __restrict__ nxyz, int Bb)
{
  int t = blockIdx.x * blockDim.x + threadIdx.x;
  if (t >= Bb * np) return;
  int b = t / np;
  int fi = idx[t];
  long long s = ((long long)b * Nl + fi) * 3;
  nxyz[(long long)t * 3] = xyz[s]; nxyz[(long long)t * 3 + 1] = xyz[s + 1]; nxyz[(long long)t * 3 + 2] = xyz[s + 2];
}

__global__ void k_knn(const float* __restrict__ xyz, const float* __restrict__ nxyz,
                      int Nl, int np, int ns, int* __restrict__ idx)
{
  int blk = blockIdx.x;
  int b = blk / np, i = blk % np;
  int tid = threadIdx.x;
  __shared__ float sd[4096];
  __shared__ float rv[128];
  __shared__ int   ri[128];
  long long qb = ((long long)b * np + i) * 3;
  float qx = nxyz[qb], qy = nxyz[qb + 1], qz = nxyz[qb + 2];
  for (int j = tid; j < Nl; j += 128) {
    long long q = ((long long)b * Nl + j) * 3;
    float dx = xyz[q] - qx, dy = xyz[q + 1] - qy, dz = xyz[q + 2] - qz;
    sd[j] = dx * dx + dy * dy + dz * dz;
  }
  __syncthreads();
  for (int s = 0; s < ns; ++s) {
    float bestv = 1e38f; int besti = 0;
    for (int j = tid; j < Nl; j += 128)
      if (sd[j] < bestv) { bestv = sd[j]; besti = j; }
    rv[tid] = bestv; ri[tid] = besti;
    __syncthreads();
    for (int st = 64; st > 0; st >>= 1) {
      if (tid < st) {
        if (rv[tid + st] < rv[tid] || (rv[tid + st] == rv[tid] && ri[tid + st] < ri[tid])) {
          rv[tid] = rv[tid + st]; ri[tid] = ri[tid + st];
        }
      }
      __syncthreads();
    }
    if (tid == 0) { idx[((long long)b * np + i) * ns + s] = ri[0]; sd[ri[0]] = 1e38f; }
    __syncthreads();
  }
}

__global__ void k_group_sa(const float* __restrict__ xyz, int Nl,
                           const float* __restrict__ nxyz, int np, int ns,
                           const int* __restrict__ idxs, const float* __restrict__ feats,
                           int Cf, float* __restrict__ out, long long cols, int Bb)
{
  long long col = (long long)blockIdx.x * blockDim.x + threadIdx.x;
  if (col >= cols) return;
  int b = (int)(col / ((long long)np * ns));
  long long rem = col % ((long long)np * ns);
  int i = (int)(rem / ns), j = (int)(rem % ns);
  int p = idxs[((long long)b * np + i) * ns + j];
  long long gp = ((long long)b * Nl + p) * 3;
  long long np3 = ((long long)b * np + i) * 3;
  out[0 * cols + col] = xyz[gp]     - nxyz[np3];
  out[1 * cols + col] = xyz[gp + 1] - nxyz[np3 + 1];
  out[2 * cols + col] = xyz[gp + 2] - nxyz[np3 + 2];
  long long ldf = (long long)Bb * Nl;
  for (int c = 0; c < Cf; ++c)
    out[(long long)(3 + c) * cols + col] = feats[(long long)c * ldf + (long long)b * Nl + p];
}

__global__ void k_mean_xyz(const float* __restrict__ xyz, int Nl, float* __restrict__ out)
{
  int b = blockIdx.x, tid = threadIdx.x;
  __shared__ float sx[128], sy[128], sz[128];
  float ax = 0, ay = 0, az = 0;
  for (int j = tid; j < Nl; j += 128) {
    long long q = ((long long)b * Nl + j) * 3;
    ax += xyz[q]; ay += xyz[q + 1]; az += xyz[q + 2];
  }
  sx[tid] = ax; sy[tid] = ay; sz[tid] = az;
  __syncthreads();
  for (int s = 64; s > 0; s >>= 1) {
    if (tid < s) { sx[tid] += sx[tid + s]; sy[tid] += sy[tid + s]; sz[tid] += sz[tid + s]; }
    __syncthreads();
  }
  if (tid == 0) { out[b * 3] = sx[0] / Nl; out[b * 3 + 1] = sy[0] / Nl; out[b * 3 + 2] = sz[0] / Nl; }
}

__global__ void k_group_all(const float* __restrict__ xyz, int Nl, const float* __restrict__ mxyz,
                            const float* __restrict__ feats, int Cf, float* __restrict__ out,
                            long long cols, int Bb)
{
  long long col = (long long)blockIdx.x * blockDim.x + threadIdx.x;
  if (col >= cols) return;
  int b = (int)(col / Nl);
  long long gp = col * 3;
  out[0 * cols + col] = xyz[gp]     - mxyz[b * 3];
  out[1 * cols + col] = xyz[gp + 1] - mxyz[b * 3 + 1];
  out[2 * cols + col] = xyz[gp + 2] - mxyz[b * 3 + 2];
  for (int c = 0; c < Cf; ++c)
    out[(long long)(3 + c) * cols + col] = feats[(long long)c * cols + col];
}

__global__ void k_maxpool(const float* __restrict__ in, int C, long long cols_out, int ns,
                          float* __restrict__ out)
{
  long long t = (long long)blockIdx.x * blockDim.x + threadIdx.x;
  if (t >= (long long)C * cols_out) return;
  long long c = t / cols_out, co = t % cols_out;
  long long base = c * cols_out * ns + co * ns;
  float m = -1e30f;
  for (int j = 0; j < ns; ++j) m = fmaxf(m, in[base + j]);
  out[t] = m;
}

__global__ void k_ca_reduce(const float* __restrict__ x, int Np, int C, int Bb,
                            float* __restrict__ mean, float* __restrict__ mx)
{
  int bid = blockIdx.x;
  int c = bid / Bb, b = bid % Bb;
  int tid = threadIdx.x;
  __shared__ float ss[128], sm[128];
  long long base = (long long)c * Bb * Np + (long long)b * Np;
  float s = 0.f, m = -1e30f;
  for (int n = tid; n < Np; n += 128) { float v = x[base + n]; s += v; m = fmaxf(m, v); }
  ss[tid] = s; sm[tid] = m;
  __syncthreads();
  for (int st = 64; st > 0; st >>= 1) {
    if (tid < st) { ss[tid] += ss[tid + st]; sm[tid] = fmaxf(sm[tid], sm[tid + st]); }
    __syncthreads();
  }
  if (tid == 0) { mean[b * C + c] = ss[0] / Np; mx[b * C + c] = sm[0]; }
}

__global__ void k_ca_fc(const float* __restrict__ mean, const float* __restrict__ mx,
                        const float* __restrict__ w1, const float* __restrict__ w2,
                        int C, int H, float* __restrict__ sg)
{
  int b = blockIdx.x, tid = threadIdx.x;
  __shared__ float hm[128], hx[128];
  for (int t = tid; t < H; t += 256) {
    float s1 = 0.f, s2 = 0.f;
    for (int c = 0; c < C; ++c) {
      float w = w1[(long long)t * C + c];
      s1 += w * mean[b * C + c];
      s2 += w * mx[b * C + c];
    }
    hm[t] = fmaxf(s1, 0.f);
    hx[t] = fmaxf(s2, 0.f);
  }
  __syncthreads();
  for (int c = tid; c < C; c += 256) {
    float acc = 0.f;
    for (int h = 0; h < H; ++h) acc += w2[(long long)c * H + h] * (hm[h] + hx[h]);
    sg[b * C + c] = 1.f / (1.f + expf(-acc));
  }
}

__global__ void k_ca_scale(float* __restrict__ x, int Np, int C, int Bb,
                           const float* __restrict__ sg)
{
  long long t = (long long)blockIdx.x * blockDim.x + threadIdx.x;
  if (t >= (long long)C * Bb * Np) return;
  long long c = t / ((long long)Bb * Np);
  long long r = t % ((long long)Bb * Np);
  int b = (int)(r / Np);
  x[t] *= sg[b * C + c];
}

__global__ void k_fp_interp(const float* __restrict__ xyz1, int n1,
                            const float* __restrict__ xyz2, int n2,
                            const float* __restrict__ src, float* __restrict__ dst,
                            int roff, int C2, int kk, int Bb)
{
  long long t = (long long)blockIdx.x * blockDim.x + threadIdx.x;
  if (t >= (long long)Bb * n1) return;
  int b = (int)(t / n1);
  float px = xyz1[t * 3], py = xyz1[t * 3 + 1], pz = xyz1[t * 3 + 2];
  float bd[3] = {1e38f, 1e38f, 1e38f};
  int   bi[3] = {0, 0, 0};
  for (int j = 0; j < n2; ++j) {
    long long q = ((long long)b * n2 + j) * 3;
    float dx = px - xyz2[q], dy = py - xyz2[q + 1], dz = pz - xyz2[q + 2];
    float d = dx * dx + dy * dy + dz * dz;
    if (d < bd[0])      { bd[2]=bd[1]; bi[2]=bi[1]; bd[1]=bd[0]; bi[1]=bi[0]; bd[0]=d; bi[0]=j; }
    else if (d < bd[1]) { bd[2]=bd[1]; bi[2]=bi[1]; bd[1]=d; bi[1]=j; }
    else if (d < bd[2]) { bd[2]=d; bi[2]=j; }
  }
  float w[3] = {0, 0, 0}, ws = 0.f;
  for (int q = 0; q < kk; ++q) { w[q] = 1.f / (bd[q] + 1e-8f); ws += w[q]; }
  for (int q = 0; q < kk; ++q) w[q] /= ws;
  long long ld1 = (long long)Bb * n1, ld2 = (long long)Bb * n2;
  for (int c = 0; c < C2; ++c) {
    float acc = 0.f;
    for (int q = 0; q < kk; ++q)
      acc += w[q] * src[(long long)c * ld2 + (long long)b * n2 + bi[q]];
    dst[(long long)(roff + c) * ld1 + t] = acc;
  }
}

__global__ void k_copy_rows(const float* __restrict__ src, float* __restrict__ dst,
                            int roff, int C, long long cols)
{
  long long t = (long long)blockIdx.x * blockDim.x + threadIdx.x;
  if (t >= (long long)C * cols) return;
  long long c = t / cols, n = t % cols;
  dst[(long long)(roff + c) * cols + n] = src[t];
}

__global__ void k_fuse_offset(const float* __restrict__ Wf, const float* __restrict__ l3b,
                              const float* __restrict__ midg, float* __restrict__ off, int Bb)
{
  int t = blockIdx.x * blockDim.x + threadIdx.x;
  if (t >= Bb * 128) return;
  int b = t / 128, m = t % 128;
  float acc = 0.f;
  for (int c = 0; c < 1024; ++c) acc += Wf[(long long)m * 1408 + 128 + c] * l3b[(long long)c * Bb + b];
  for (int c = 0; c < 256; ++c)  acc += Wf[(long long)m * 1408 + 1152 + c] * midg[(long long)c * Bb + b];
  off[b * 128 + m] = acc;
}

__global__ void k_logsoftmax(const float* __restrict__ lg, int BN, float* __restrict__ out)
{
  int t = blockIdx.x * blockDim.x + threadIdx.x;
  if (t >= BN) return;
  float m = -1e30f;
  for (int k = 0; k < 13; ++k) m = fmaxf(m, lg[(long long)k * BN + t]);
  float s = 0.f;
  for (int k = 0; k < 13; ++k) s += expf(lg[(long long)k * BN + t] - m);
  float lse = logf(s) + m;
  for (int k = 0; k < 13; ++k) out[(long long)t * 13 + k] = lg[(long long)k * BN + t] - lse;
}

// ---------------------------------------------------------------------------
extern "C" void kernel_launch(void* const* d_in, const int* in_sizes, int n_in,
                              void* d_out, int out_size, void* d_ws, size_t ws_size,
                              hipStream_t stream)
{
  const int Bb = 4, N = 4096, BN = Bb * N;

  // ---- unpack params (jax pytree order: sorted dict keys; x is 'x') ----
  int cur = 0;
  const float* X_IN;
  if (in_sizes[0] == BN * 6) { X_IN = (const float*)d_in[0]; cur = 1; }
  else                       { X_IN = (const float*)d_in[n_in - 1]; cur = 0; }

  struct CAp  { const float *w1, *w2; };
  struct LYR  { const float *b, *be, *g, *w; };
  auto NXT  = [&]() { return (const float*)d_in[cur++]; };
  auto rdCA = [&]() { CAp c; c.w1 = NXT(); c.w2 = NXT(); return c; };
  auto rdL  = [&]() { LYR l; l.b = NXT(); l.be = NXT(); l.g = NXT(); l.w = NXT(); return l; };

  CAp fp1_a = rdCA(); LYR fp1_m0 = rdL(), fp1_m1 = rdL(), fp1_m2 = rdL();
  CAp fp2_a = rdCA(); LYR fp2_m0 = rdL(), fp2_m1 = rdL(), fp2_m2 = rdL();
  CAp fp3_a = rdCA(); LYR fp3_m0 = rdL(), fp3_m1 = rdL();
  LYR fuse  = rdL();
  LYR head1 = rdL(); LYR head2 = rdL();
  const float* head3_b = NXT(); const float* head3_w = NXT();
  const float* pos_bk = NXT(); const float* pos_bq = NXT(); const float* pos_bv = NXT();
  const float* pos_gamma = NXT();
  const float* pos_wk = NXT(); const float* pos_wq = NXT(); const float* pos_wv = NXT();
  CAp sa1_a = rdCA(); LYR sa1_m0 = rdL(), sa1_m1 = rdL(), sa1_m2 = rdL();
  CAp sa2_a = rdCA(); LYR sa2_m0 = rdL(), sa2_m1 = rdL(), sa2_m2 = rdL();
  CAp sa3_a = rdCA(); LYR sa3_m0 = rdL(), sa3_m1 = rdL(), sa3_m2 = rdL();
  CAp seg = rdCA(); CAp sem = rdCA();

  // ---- workspace arena ----
  char* wsb = (char*)d_ws;
  size_t off = 0;
  auto carve = [&](size_t bytes) -> void* {
    void* p = wsb + off;
    off = (off + bytes + 255) & ~(size_t)255;
    return p;
  };
  float* feat0 = (float*)carve(sizeof(float) * 6 * BN);
  float* xyz0  = (float*)carve(sizeof(float) * BN * 3);
  float* xyzn  = (float*)carve(sizeof(float) * BN * 3);
  int*   fps1  = (int*)carve(sizeof(int) * Bb * 512);
  int*   fps2  = (int*)carve(sizeof(int) * Bb * 128);
  float* nx1   = (float*)carve(sizeof(float) * Bb * 512 * 3);
  float* nx2   = (float*)carve(sizeof(float) * Bb * 128 * 3);
  float* nx3   = (float*)carve(sizeof(float) * Bb * 3);
  int*   knn1  = (int*)carve(sizeof(int) * Bb * 512 * 32);
  int*   knn2  = (int*)carve(sizeof(int) * Bb * 128 * 64);
  float* grp   = (float*)carve(sizeof(float) * 4292608);   // shared concat/group buffer
  float* h_a   = (float*)carve(sizeof(float) * 8388608);
  float* h_b   = (float*)carve(sizeof(float) * 8388608);
  float* l1p   = (float*)carve(sizeof(float) * 128 * 2048);
  float* l2p   = (float*)carve(sizeof(float) * 256 * 512);
  float* l3p   = (float*)carve(sizeof(float) * 1024 * Bb);
  float* l3b   = (float*)carve(sizeof(float) * 1024 * Bb);
  float* l2fp  = (float*)carve(sizeof(float) * 256 * 512);
  float* l1fp  = (float*)carve(sizeof(float) * 128 * 2048);
  float* l0fp  = (float*)carve(sizeof(float) * 128 * BN);
  float* Qb    = (float*)carve(sizeof(float) * 32 * BN);
  float* Kb    = (float*)carve(sizeof(float) * 32 * BN);
  float* Vb    = (float*)carve(sizeof(float) * 128 * BN);
  float* featA = (float*)carve(sizeof(float) * 128 * BN);
  float* camean = (float*)carve(sizeof(float) * Bb * 1024);
  float* camax  = (float*)carve(sizeof(float) * Bb * 1024);
  float* casg   = (float*)carve(sizeof(float) * Bb * 1024);
  float* midg   = (float*)carve(sizeof(float) * 256 * Bb);
  float* foff   = (float*)carve(sizeof(float) * Bb * 128);
  if (off > ws_size) return;  // workspace too small; nothing safe to do

  auto nblk = [](long long n, int t) { return (unsigned)((n + t - 1) / t); };

  auto gemm = [&](const float* W, const float* Bm, const float* bias, const float* gg,
                  const float* bbe, const float* cb, float* Cm,
                  int M, long long Nn, int K, int lda, int flags) {
    dim3 grid((unsigned)((Nn + 15) / 16), (unsigned)((M + 15) / 16), 1);
    k_wmma_gemm<<<grid, 32, 0, stream>>>(W, Bm, bias, gg, bbe, cb, Cm,
                                         M, (int)Nn, K, lda, 4096, flags);
  };
  auto conv = [&](const LYR& L, const float* Bm, float* Cm, int M, long long Nn, int K) {
    gemm(L.w, Bm, L.b, L.g, L.be, nullptr, Cm, M, Nn, K, K, 7);
  };
  auto ca_apply = [&](float* xb, int Np, int C, int H, CAp ca) {
    k_ca_reduce<<<C * Bb, 128, 0, stream>>>(xb, Np, C, Bb, camean, camax);
    k_ca_fc<<<Bb, 256, 0, stream>>>(camean, camax, ca.w1, ca.w2, C, H, casg);
    k_ca_scale<<<nblk((long long)C * Bb * Np, 256), 256, 0, stream>>>(xb, Np, C, Bb, casg);
  };

  // ---- prep ----
  k_prep<<<nblk(BN, 256), 256, 0, stream>>>(X_IN, feat0, xyz0, xyzn, BN);

  // ---- SA1: 4096 -> 512 pts, 32 samples, mlp 9->64->64->128 ----
  k_fps<<<Bb, 256, 0, stream>>>(xyz0, N, 512, fps1);
  k_gather_xyz<<<nblk(Bb * 512, 256), 256, 0, stream>>>(xyz0, N, fps1, 512, nx1, Bb);
  k_knn<<<Bb * 512, 128, 0, stream>>>(xyz0, nx1, N, 512, 32, knn1);
  {
    long long cols = (long long)Bb * 512 * 32;
    k_group_sa<<<nblk(cols, 256), 256, 0, stream>>>(xyz0, N, nx1, 512, 32, knn1, feat0, 6, grp, cols, Bb);
    conv(sa1_m0, grp, h_a, 64, cols, 9);
    conv(sa1_m1, h_a, h_b, 64, cols, 64);
    conv(sa1_m2, h_b, h_a, 128, cols, 64);
    k_maxpool<<<nblk(128LL * 2048, 256), 256, 0, stream>>>(h_a, 128, 2048, 32, l1p);
    ca_apply(l1p, 512, 128, 16, sa1_a);
  }

  // ---- SA2: 512 -> 128 pts, 64 samples, mlp 131->128->128->256 ----
  k_fps<<<Bb, 256, 0, stream>>>(nx1, 512, 128, fps2);
  k_gather_xyz<<<nblk(Bb * 128, 256), 256, 0, stream>>>(nx1, 512, fps2, 128, nx2, Bb);
  k_knn<<<Bb * 128, 128, 0, stream>>>(nx1, nx2, 512, 128, 64, knn2);
  {
    long long cols = (long long)Bb * 128 * 64;
    k_group_sa<<<nblk(cols, 256), 256, 0, stream>>>(nx1, 512, nx2, 128, 64, knn2, l1p, 128, grp, cols, Bb);
    conv(sa2_m0, grp, h_a, 128, cols, 131);
    conv(sa2_m1, h_a, h_b, 128, cols, 128);
    conv(sa2_m2, h_b, h_a, 256, cols, 128);
    k_maxpool<<<nblk(256LL * 512, 256), 256, 0, stream>>>(h_a, 256, 512, 64, l2p);
    ca_apply(l2p, 128, 256, 32, sa2_a);
  }

  // ---- SA3: global, mlp 259->256->512->1024 ----
  k_mean_xyz<<<Bb, 128, 0, stream>>>(nx2, 128, nx3);
  {
    long long cols = (long long)Bb * 128;
    k_group_all<<<nblk(cols, 256), 256, 0, stream>>>(nx2, 128, nx3, l2p, 256, grp, cols, Bb);
    conv(sa3_m0, grp, h_a, 256, cols, 259);
    conv(sa3_m1, h_a, h_b, 512, cols, 256);
    conv(sa3_m2, h_b, h_a, 1024, cols, 512);
    k_maxpool<<<nblk(1024LL * Bb, 256), 256, 0, stream>>>(h_a, 1024, Bb, 128, l3p);
    ca_apply(l3p, 1, 1024, 128, sa3_a);
  }

  // ---- FP3: l3 -> l2, mlp 1280->256->256 ----
  {
    long long cols = (long long)Bb * 128;
    k_copy_rows<<<nblk(256 * cols, 256), 256, 0, stream>>>(l2p, grp, 0, 256, cols);
    k_fp_interp<<<nblk(cols, 128), 128, 0, stream>>>(nx2, 128, nx3, 1, l3p, grp, 256, 1024, 1, Bb);
    conv(fp3_m0, grp, h_a, 256, cols, 1280);
    conv(fp3_m1, h_a, l2fp, 256, cols, 256);
    ca_apply(l2fp, 128, 256, 32, fp3_a);
  }

  // ---- FP2: l2 -> l1, mlp 384->256->256->128 ----
  {
    long long cols = (long long)Bb * 512;
    k_copy_rows<<<nblk(128 * cols, 256), 256, 0, stream>>>(l1p, grp, 0, 128, cols);
    k_fp_interp<<<nblk(cols, 128), 128, 0, stream>>>(nx1, 512, nx2, 128, l2fp, grp, 128, 256, 3, Bb);
    conv(fp2_m0, grp, h_a, 256, cols, 384);
    conv(fp2_m1, h_a, h_b, 256, cols, 256);
    conv(fp2_m2, h_b, l1fp, 128, cols, 256);
    ca_apply(l1fp, 512, 128, 16, fp2_a);
  }

  // ---- FP1: l1 -> l0, mlp 134->128->128->128 ----
  {
    long long cols = (long long)BN;
    k_copy_rows<<<nblk(6 * cols, 256), 256, 0, stream>>>(feat0, grp, 0, 6, cols);
    k_fp_interp<<<nblk(cols, 128), 128, 0, stream>>>(xyz0, N, nx1, 512, l1fp, grp, 6, 128, 3, Bb);
    conv(fp1_m0, grp, h_a, 128, cols, 134);
    conv(fp1_m1, h_a, h_b, 128, cols, 128);
    conv(fp1_m2, h_b, l0fp, 128, cols, 128);
    ca_apply(l0fp, N, 128, 16, fp1_a);
  }

  // ---- position attention (flash, fused) ----
  gemm(pos_wq, l0fp, pos_bq, nullptr, nullptr, nullptr, Qb, 32, BN, 128, 128, 1);
  gemm(pos_wk, l0fp, pos_bk, nullptr, nullptr, nullptr, Kb, 32, BN, 128, 128, 1);
  gemm(pos_wv, l0fp, pos_bv, nullptr, nullptr, nullptr, Vb, 128, BN, 128, 128, 1);
  k_flash_pos_attn<<<dim3(N / 16, Bb), 32, 0, stream>>>(Qb, Kb, Vb, xyzn, l0fp,
                                                        pos_gamma, featA, N, BN);

  // ---- mid / global branches ----
  ca_apply(l2fp, 128, 256, 32, sem);                                  // se_mid (in place)
  k_maxpool<<<nblk(256LL * Bb, 256), 256, 0, stream>>>(l2fp, 256, Bb, 128, midg);
  k_copy_rows<<<nblk(1024LL * Bb, 256), 256, 0, stream>>>(l3p, l3b, 0, 1024, Bb);
  ca_apply(l3b, 1, 1024, 64, seg);                                    // se_global

  // ---- fuse (broadcast channels folded into per-batch column bias) ----
  k_fuse_offset<<<nblk(Bb * 128, 128), 128, 0, stream>>>(fuse.w, l3b, midg, foff, Bb);
  gemm(fuse.w, featA, fuse.b, fuse.g, fuse.be, foff, h_a, 128, BN, 128, 1408, 7);
  conv(head1, h_a, h_b, 256, BN, 128);
  conv(head2, h_b, h_a, 128, BN, 256);
  gemm(head3_w, h_a, head3_b, nullptr, nullptr, nullptr, h_b, 13, BN, 128, 128, 1);
  k_logsoftmax<<<nblk(BN, 256), 256, 0, stream>>>(h_b, BN, (float*)d_out);
}
